// HashGridVoxel_19688130084948
// MI455X (gfx1250) — matrix operand order repaired and checked by reading, not compiled, not run on hardware
//
#include <hip/hip_runtime.h>

// ---------------------------------------------------------------------------
// instant-NGP HashGrid encode + 32->64->1 MLP, fused, for gfx1250 (MI455X).
//
// Perf model: hash tables are 64 MB total -> resident in MI455X's 192 MB L2;
// the kernel is gather-bound (33.5M float2 gathers). HBM compulsory traffic is
// ~70 MB (~3us @ 23.3 TB/s). The MLP layer-1 GEMM (N x 64 x 32) is done with
// v_wmma_f32_16x16x32_f16 (f16 in / f32 accumulate), layer-2 as an in-register
// cross-lane reduction.
// ---------------------------------------------------------------------------

typedef _Float16 v16h __attribute__((ext_vector_type(16)));
typedef _Float16 v8h  __attribute__((ext_vector_type(8)));
typedef float    v8f  __attribute__((ext_vector_type(8)));

#define NGP_L      16
#define NGP_T      (1u << 19)          // hash entries per level
#define NGP_TMASK  (NGP_T - 1u)
#define ROWH       40                  // LDS row stride in halves (80B, 16B aligned)

// res_l = floor(16 * 2^(8l/15) + 1e-6); dense iff (res+1)^3 <= T  (levels 0..4)
__device__ __constant__ const int kRes[NGP_L] = {
    16, 23, 33, 48, 70, 101, 147, 212, 307, 445, 645, 933, 1351, 1955, 2830, 4096};

__global__ __launch_bounds__(256)
void ngp_hashgrid_mlp_kernel(const float* __restrict__ pts,
                             const float* __restrict__ table,
                             const float* __restrict__ w1,
                             const float* __restrict__ w2,
                             float* __restrict__ out,
                             int npts)
{
    __shared__ _Float16 encLds[256 * ROWH];   // 20 KB: 256 points x 32 f16 feats

    const int tid = threadIdx.x;
    int p = blockIdx.x * 256 + tid;
    const int pc = (p < npts) ? p : (npts - 1);   // clamp for tail safety

    const float px = pts[3 * pc + 0];
    const float py = pts[3 * pc + 1];
    const float pz = pts[3 * pc + 2];

    _Float16* encRow = &encLds[tid * ROWH];

    // ---------------- hash-grid encoding: 16 levels x 8 corners -------------
    #pragma unroll
    for (int l = 0; l < NGP_L; ++l) {
        const int   res   = kRes[l];
        const bool  dense = (l < 5);               // (res+1)^3 <= 2^19
        const float rf    = (float)res;

        const float xs = px * rf, ys = py * rf, zs = pz * rf;
        const float xf = floorf(xs), yf = floorf(ys), zf = floorf(zs);
        const float fx = xs - xf,  fy = ys - yf,  fz = zs - zf;

        const unsigned cx0 = (unsigned)xf;
        const unsigned cy0 = (unsigned)yf;
        const unsigned cz0 = (unsigned)zf;
        const unsigned r1  = (unsigned)(res + 1);

        const float wx[2] = {1.0f - fx, fx};
        const float wy[2] = {1.0f - fy, fy};
        const float wz[2] = {1.0f - fz, fz};

        const float2* __restrict__ tabL =
            (const float2*)table + (size_t)l * NGP_T;

        float a0 = 0.0f, a1 = 0.0f;
        #pragma unroll
        for (int c = 0; c < 8; ++c) {
            const unsigned ox = (c >> 2) & 1u;     // OFFS meshgrid 'ij' order
            const unsigned oy = (c >> 1) & 1u;
            const unsigned oz = c & 1u;
            const unsigned cx = cx0 + ox;
            const unsigned cy = cy0 + oy;
            const unsigned cz = cz0 + oz;

            unsigned idx;
            if (dense) idx = cx + r1 * (cy + r1 * cz);
            else       idx = (cx * 1u) ^ (cy * 2654435761u) ^ (cz * 805459861u);
            idx &= NGP_TMASK;

            const float2 fv = tabL[idx];           // global_load_b64 gather
            const float  w  = wx[ox] * wy[oy] * wz[oz];
            a0 = fmaf(w, fv.x, a0);
            a1 = fmaf(w, fv.y, a1);
        }
        encRow[2 * l + 0] = (_Float16)a0;          // level-major: k = 2l + f
        encRow[2 * l + 1] = (_Float16)a1;
    }

    __syncthreads();

    // ---------------- MLP: per-wave WMMA over 32 points ---------------------
    const int lane = tid & 31;
    const int wave = tid >> 5;
    const int nsub = lane & 15;
    const int hi   = lane >> 4;

    // B fragments (w1: [64 out][32 in], B[k][j] = w1[j][k]).
    // B layout used here (contiguous k): lanes 0-15 hold K=0..15 of column n=lane,
    //               lanes 16-31 hold K=16..31 of column n=lane-16.
    v16h bfrag[4];
    #pragma unroll
    for (int t = 0; t < 4; ++t) {
        const float4* __restrict__ wq =
            (const float4*)(w1 + (size_t)(t * 16 + nsub) * 32 + hi * 16);
        float4 q0 = wq[0], q1 = wq[1], q2 = wq[2], q3 = wq[3];
        bfrag[t][0]  = (_Float16)q0.x; bfrag[t][1]  = (_Float16)q0.y;
        bfrag[t][2]  = (_Float16)q0.z; bfrag[t][3]  = (_Float16)q0.w;
        bfrag[t][4]  = (_Float16)q1.x; bfrag[t][5]  = (_Float16)q1.y;
        bfrag[t][6]  = (_Float16)q1.z; bfrag[t][7]  = (_Float16)q1.w;
        bfrag[t][8]  = (_Float16)q2.x; bfrag[t][9]  = (_Float16)q2.y;
        bfrag[t][10] = (_Float16)q2.z; bfrag[t][11] = (_Float16)q2.w;
        bfrag[t][12] = (_Float16)q3.x; bfrag[t][13] = (_Float16)q3.y;
        bfrag[t][14] = (_Float16)q3.z; bfrag[t][15] = (_Float16)q3.w;
    }

    float w2v[4];
    #pragma unroll
    for (int t = 0; t < 4; ++t) w2v[t] = w2[t * 16 + nsub];

    #pragma unroll
    for (int tt = 0; tt < 2; ++tt) {               // two 16-point M-tiles
        const int row = wave * 32 + tt * 16 + nsub;

        // A fragment (16x32 f16). Layout: lane m: halves 0-15 = K 16*hi..16*hi+15
        // (matches B); our LDS row is 16B-aligned per chunk.
        const _Float16* rp = &encLds[row * ROWH];
        const v8h alo = *(const v8h*)(rp + 16 * hi);
        const v8h ahi = *(const v8h*)(rp + 16 * hi + 8);
        v16h a;
        #pragma unroll
        for (int i = 0; i < 8; ++i) { a[i] = alo[i]; a[8 + i] = ahi[i]; }

        v8f c[4] = {v8f{}, v8f{}, v8f{}, v8f{}};
        #pragma unroll
        for (int t = 0; t < 4; ++t) {
            c[t] = __builtin_amdgcn_wmma_f32_16x16x32_f16(
                false, a, false, bfrag[t], (short)0, c[t], false, false);
        }

        // Layer 2: out = sigmoid( sum_j relu(h[j]) * w2[j] ).
        // C layout: VGPR r, lane l -> h[m = r + 8*(l>>4)][j = 16t + (l&15)].
        float myval = 0.0f;
        #pragma unroll
        for (int r = 0; r < 8; ++r) {
            float partial = fmaxf(c[0][r], 0.0f) * w2v[0]
                          + fmaxf(c[1][r], 0.0f) * w2v[1]
                          + fmaxf(c[2][r], 0.0f) * w2v[2]
                          + fmaxf(c[3][r], 0.0f) * w2v[3];
            partial += __shfl_xor(partial, 1);
            partial += __shfl_xor(partial, 2);
            partial += __shfl_xor(partial, 4);
            partial += __shfl_xor(partial, 8);     // sum over 16 lanes per half
            if (nsub == r) myval = partial;
        }

        if (nsub < 8) {
            const int op = blockIdx.x * 256 + wave * 32 + tt * 16 + nsub + 8 * hi;
            if (op < npts)
                out[op] = 1.0f / (1.0f + __expf(-myval));
        }
    }
}

extern "C" void kernel_launch(void* const* d_in, const int* in_sizes, int n_in,
                              void* d_out, int out_size, void* d_ws, size_t ws_size,
                              hipStream_t stream) {
    const float* pts   = (const float*)d_in[0];   // [N,3] f32
    const float* table = (const float*)d_in[1];   // [16, 2^19, 2] f32
    const float* w1    = (const float*)d_in[2];   // [64,32] f32
    const float* w2    = (const float*)d_in[3];   // [1,64] f32
    float* out = (float*)d_out;                   // [N] f32

    const int npts   = in_sizes[0] / 3;           // 262144
    const int blocks = (npts + 255) / 256;
    ngp_hashgrid_mlp_kernel<<<dim3(blocks), dim3(256), 0, stream>>>(
        pts, table, w1, w2, out, npts);
}
